// MultiScaleDeformableAttention_60842506715751
// MI455X (gfx1250) — matrix operand order repaired and checked by reading, not compiled, not run on hardware
//
#include <hip/hip_runtime.h>
#include <hip/hip_bf16.h>
#include <cstddef>
#include <cstdint>

// Problem constants (match reference)
#define NHEAD 8
#define HDIM  32
#define NLVL  4
#define NPTS  4
#define BB    4
#define QQ    1024
#define DD    256
#define SS    21760

typedef __attribute__((ext_vector_type(16))) __bf16 v16bf;
typedef __attribute__((ext_vector_type(8)))  __bf16 v8bf;
typedef __attribute__((ext_vector_type(2)))  __bf16 v2bf;
typedef __attribute__((ext_vector_type(8)))  float  v8f;

typedef __attribute__((ext_vector_type(4))) unsigned int u32x4;
typedef __attribute__((ext_vector_type(8))) int          i32x8;
typedef __attribute__((ext_vector_type(4))) int          i32x4;

#if __has_builtin(__builtin_amdgcn_tensor_load_to_lds)
#define HAVE_TDM 1
#else
#define HAVE_TDM 0
#endif

static __device__ __forceinline__ unsigned short f2b(float f) {
    unsigned u = __float_as_uint(f);
    u += 0x7FFFu + ((u >> 16) & 1u);     // round-to-nearest-even bf16
    return (unsigned short)(u >> 16);
}
static __device__ __forceinline__ float b2f(unsigned short h) {
    return __uint_as_float(((unsigned)h) << 16);
}

#if HAVE_TDM
// ---------------------------------------------------------------------------
// Issue a TDM 2D tile load: global (row-major, row_stride elems of 2B)
// -> LDS at lds_addr, tile_x elems per row, tile_y rows, with LDS row padding
// (pad_interval_code: pad every 2^(c+1) DWORDs; pad_amount_code: c+1 DWORDs).
// Descriptor layout per CDNA5 ISA. 2D tile ->
// groups 2/3 zero. Uniform (SGPR) operands; EXEC is ignored by TDM.
// ---------------------------------------------------------------------------
static __device__ __forceinline__ void tdm_load_2d(
    unsigned lds_addr, const void* gptr,
    unsigned tile_x, unsigned tile_y, unsigned row_stride_elems,
    unsigned pad_interval_code, unsigned pad_amount_code)
{
    const unsigned long long ga = (unsigned long long)gptr;
    u32x4 g0;
    g0[0] = 1u;                                        // count=1 (valid user D#)
    g0[1] = lds_addr;                                  // lds_addr [63:32]
    g0[2] = (unsigned)(ga & 0xFFFFFFFFu);              // global_addr[31:0]
    g0[3] = ((unsigned)(ga >> 32) & 0x01FFFFFFu)       // global_addr[56:32]
          | 0x80000000u;                               // type=2 ("image")
    const unsigned tensor_x = 1u << 30;                // huge -> OOB never fires
    const unsigned tensor_y = 1u << 30;
    i32x8 g1;
    g1[0] = (int)((1u << 16)                           // data_size = 2 bytes
                | (1u << 20)                           // pad_enable
                | (pad_interval_code << 22)
                | (pad_amount_code << 25));
    g1[1] = (int)((tensor_x & 0xFFFFu) << 16);         // tensor_dim0[15:0]
    g1[2] = (int)((tensor_x >> 16) | ((tensor_y & 0xFFFFu) << 16));
    g1[3] = (int)((tensor_y >> 16) | (tile_x << 16));  // tile_dim0
    g1[4] = (int)(tile_y & 0xFFFFu);                   // tile_dim1; tile_dim2=0
    g1[5] = (int)row_stride_elems;                     // tensor_dim0_stride[31:0]
    g1[6] = 0;
    g1[7] = 0;
    const i32x4 z4 = {0, 0, 0, 0};
#if defined(__clang_major__) && (__clang_major__ >= 23)
    const i32x8 z8 = {0, 0, 0, 0, 0, 0, 0, 0};
    __builtin_amdgcn_tensor_load_to_lds(g0, g1, z4, z4, z8, 0);
#else
    __builtin_amdgcn_tensor_load_to_lds(g0, g1, z4, z4, 0);
#endif
}
#endif // HAVE_TDM

// -------------------------------------------------------------------------
// Weight transpose + f32->bf16 convert:  out[n*K + k] = bf16(in[k*N + n])
// -------------------------------------------------------------------------
__global__ __launch_bounds__(256) void wcvt_t(const float* __restrict__ in,
                                              unsigned short* __restrict__ out,
                                              int K, int N) {
    int t = blockIdx.x * 256 + threadIdx.x;
    if (t >= K * N) return;
    int k = t / N, n = t - k * N;
    out[(size_t)n * K + k] = f2b(in[t]);
}

// -------------------------------------------------------------------------
// bf16 WMMA GEMM:  C[M,N] = A[M,K] * B[K,N] + bias[N]
//   B pre-transposed bf16 (Bt[n*K+k]), staged to LDS by the Tensor Data
//   Mover (padded rows).  A is f32 (converted to bf16 while staging, loads
//   batched so all four B128 loads are in flight) or bf16 (TDM-staged).
//   Block tile 128x64, 8 waves (4x2), wave tile 32x32 via 2x2
//   v_wmma_f32_16x16x32_bf16.  M%128==N%64==K%32==0.
// EPI==0: scatter bf16 into v layout [B,NH,S,HD];  EPI==1: f32 row-major.
// -------------------------------------------------------------------------
template<int EPI, bool ABF16>
__global__ __launch_bounds__(256) void gemm_bf16(
    const void* __restrict__ Aip, const unsigned short* __restrict__ Bt,
    const float* __restrict__ bias, void* __restrict__ outp,
    int M, int N, int K, int ldout)
{
    (void)M; (void)N;
    __shared__ unsigned short ldsA[128 * 40];   // 128 x 32 bf16, row stride 40
    __shared__ unsigned short ldsB[64 * 48];    // 64  x 32 bf16, row stride 48

    const int tid  = threadIdx.x;
    const int lane = tid & 31;
    const int wid  = tid >> 5;
    const int wm   = wid & 3;     // wave row (4)
    const int wn   = wid >> 2;    // wave col (2)
    const int m0   = blockIdx.x * 128;
    const int n0   = blockIdx.y * 64;

    v8f acc[2][2];
    #pragma unroll
    for (int r = 0; r < 2; ++r)
        #pragma unroll
        for (int c = 0; c < 2; ++c)
            #pragma unroll
            for (int e = 0; e < 8; ++e) acc[r][c][e] = 0.f;

    for (int k0 = 0; k0 < K; k0 += 32) {
        // ---- stage A tile (128x32) f32 -> bf16 in LDS (cooperative) ----
        if (!ABF16) {
            const float* Af = (const float*)Aip;
            float4 t[4];
            #pragma unroll
            for (int i = 0; i < 4; ++i) {            // issue all 4 loads first
                int li  = tid + i * 256;             // 0..1023
                int row = li >> 3;                   // 0..127
                int c4  = (li & 7) << 2;             // 0,4,...,28
                t[i] = *(const float4*)(Af + (size_t)(m0 + row) * K + k0 + c4);
            }
            #pragma unroll
            for (int i = 0; i < 4; ++i) {            // then convert + store
                int li  = tid + i * 256;
                int row = li >> 3;
                int c4  = (li & 7) << 2;
                unsigned short* d = &ldsA[row * 40 + c4];
#if __has_builtin(__builtin_amdgcn_cvt_pk_bf16_f32)
                *(v2bf*)&d[0] = __builtin_amdgcn_cvt_pk_bf16_f32(t[i].x, t[i].y);
                *(v2bf*)&d[2] = __builtin_amdgcn_cvt_pk_bf16_f32(t[i].z, t[i].w);
#else
                d[0] = f2b(t[i].x); d[1] = f2b(t[i].y);
                d[2] = f2b(t[i].z); d[3] = f2b(t[i].w);
#endif
            }
        }

#if HAVE_TDM
        // ---- B tile (and bf16 A tile) via Tensor Data Mover ----
        if (wid == 0) {
            tdm_load_2d((unsigned)(unsigned long long)(void*)&ldsB[0],
                        Bt + (size_t)n0 * K + k0,
                        /*tile_x=*/32, /*tile_y=*/64, /*stride=*/K,
                        /*interval(16 dw)=*/3, /*amount(8 dw)=*/7);
            if (ABF16)
                tdm_load_2d((unsigned)(unsigned long long)(void*)&ldsA[0],
                            (const unsigned short*)Aip + (size_t)m0 * K + k0,
                            /*tile_x=*/32, /*tile_y=*/128, /*stride=*/K,
                            /*interval(16 dw)=*/3, /*amount(4 dw)=*/3);
            __builtin_amdgcn_s_wait_tensorcnt(0);
        }
#else
        {
            int row = tid >> 2;                   // 0..63
            int c8  = (tid & 3) << 3;             // 0,8,16,24
            *(uint4*)&ldsB[row * 48 + c8] =
                *(const uint4*)(Bt + (size_t)(n0 + row) * K + k0 + c8);
        }
        if (ABF16) {
            const unsigned short* Ab = (const unsigned short*)Aip;
            #pragma unroll
            for (int i = 0; i < 4; ++i) {
                int li  = tid + i * 256;
                int row = li >> 3;
                int c4  = (li & 7) << 2;
                *(ushort4*)&ldsA[row * 40 + c4] =
                    *(const ushort4*)(Ab + (size_t)(m0 + row) * K + k0 + c4);
            }
        }
#endif
        __syncthreads();

        // ---- fragments (16-bit A/B, contiguous k = half*16 + e in both) ----
        {
            const int lm = lane & 15, half = lane >> 4;
            v16bf afr[2], bfr[2];
            #pragma unroll
            for (int r = 0; r < 2; ++r) {
                const unsigned short* ap = &ldsA[(wm * 32 + r * 16 + lm) * 40];
                v8bf lo = *(const v8bf*)(ap + half * 16);       // K = half*16 .. +7
                v8bf hi = *(const v8bf*)(ap + half * 16 + 8);   // K = half*16+8 .. +7
                afr[r] = __builtin_shufflevector(lo, hi,
                          0,1,2,3,4,5,6,7,8,9,10,11,12,13,14,15);
            }
            #pragma unroll
            for (int c = 0; c < 2; ++c)   // lane n holds column n: K = e + half*16
                bfr[c] = *(const v16bf*)&ldsB[(wn * 32 + c * 16 + lm) * 48 + half * 16];

            #pragma unroll
            for (int r = 0; r < 2; ++r)
                #pragma unroll
                for (int c = 0; c < 2; ++c)
                    acc[r][c] = __builtin_amdgcn_wmma_f32_16x16x32_bf16(
                        false, afr[r], false, bfr[c], (short)0, acc[r][c], false, false);
        }
        __syncthreads();
    }

    // ---- epilogue: C layout lane(n,half), element e -> row e + half*8 ----
    const int lm = lane & 15, half = lane >> 4;
    #pragma unroll
    for (int r = 0; r < 2; ++r) {
        #pragma unroll
        for (int c = 0; c < 2; ++c) {
            const int colg = n0 + wn * 32 + c * 16 + lm;
            const float bb = bias[colg];
            #pragma unroll
            for (int e = 0; e < 8; ++e) {
                const int mg  = m0 + wm * 32 + r * 16 + half * 8 + e;
                const float v = acc[r][c][e] + bb;
                if (EPI == 0) {
                    // scatter into v_bf16 [B, NH, S, HD]
                    const int bidx = mg / SS;
                    const int s    = mg - bidx * SS;
                    const int hh   = colg >> 5;
                    const int dd_  = colg & 31;
                    ((unsigned short*)outp)
                        [((size_t)(bidx * NHEAD + hh) * SS + s) * HDIM + dd_] = f2b(v);
                } else {
                    ((float*)outp)[(size_t)mg * ldout + colg] = v;
                }
            }
        }
    }
}

// -------------------------------------------------------------------------
// Fused softmax + multi-scale bilinear sampling.
// One wave per (b,q,head); lane = channel d (HD==32==warpSize).
// Gathers from v_bf16 [B,NH,S,HD] (L2-resident, 64B coalesced per corner).
// Writes sampled features bf16 [B*Q, NH*HD] (A-matrix of the output GEMM).
// -------------------------------------------------------------------------
__global__ __launch_bounds__(256) void msda_sample(
    const float* __restrict__ refp, const float* __restrict__ off,
    const float* __restrict__ logits, const unsigned short* __restrict__ vbf,
    unsigned short* __restrict__ sampled)
{
    const int lane = threadIdx.x & 31;
    const int wg   = blockIdx.x * 8 + (threadIdx.x >> 5);
    const int h    = wg & (NHEAD - 1);
    const int bq   = wg >> 3;
    const int b    = bq >> 10;          // QQ = 1024
    const int q    = bq & (QQ - 1);

    // softmax over 16 logits (redundant per lane; cheap VALU)
    const float* lg = logits + (size_t)bq * (NHEAD * NLVL * NPTS) + h * (NLVL * NPTS);
    float l[16], mx = -1e30f;
    #pragma unroll
    for (int i = 0; i < 16; ++i) { l[i] = lg[i]; mx = fmaxf(mx, l[i]); }
    float sum = 0.f;
    #pragma unroll
    for (int i = 0; i < 16; ++i) { l[i] = __expf(l[i] - mx); sum += l[i]; }
    const float inv = 1.f / sum;

    const float* rp = refp + (size_t)(b * QQ + q) * NLVL * 2;
    const float* op = off + (size_t)bq * (NHEAD * NLVL * NPTS * 2) + h * (NLVL * NPTS * 2);

    const int lvlW[4] = {128, 64, 32, 16};
    const int lvlH[4] = {128, 64, 32, 16};
    const int lvlS[4] = {0, 16384, 20480, 21504};

    float acc = 0.f;
    #pragma unroll
    for (int lv = 0; lv < NLVL; ++lv) {
        const int   Wl = lvlW[lv], Hl = lvlH[lv];
        const float rx = rp[lv * 2 + 0], ry = rp[lv * 2 + 1];
        const size_t base = ((size_t)(b * NHEAD + h) * SS + lvlS[lv]) * HDIM;
        #pragma unroll
        for (int p = 0; p < NPTS; ++p) {
            const float lx = rx + op[(lv * NPTS + p) * 2 + 0] / (float)Wl;
            const float ly = ry + op[(lv * NPTS + p) * 2 + 1] / (float)Hl;
            const float x = lx * (float)Wl - 0.5f;
            const float y = ly * (float)Hl - 0.5f;
            const float x0f = floorf(x), y0f = floorf(y);
            const float wx = x - x0f, wy = y - y0f;
            const int x0 = (int)x0f, y0 = (int)y0f;
            const float aw = l[lv * NPTS + p] * inv;
            #pragma unroll
            for (int dy = 0; dy < 2; ++dy) {
                #pragma unroll
                for (int dx = 0; dx < 2; ++dx) {
                    const int xi = x0 + dx, yi = y0 + dy;
                    if (xi >= 0 && xi < Wl && yi >= 0 && yi < Hl) {
                        const float w = (dy ? wy : 1.f - wy) * (dx ? wx : 1.f - wx);
                        const float vv =
                            b2f(vbf[base + (size_t)(yi * Wl + xi) * HDIM + lane]);
                        acc += aw * w * vv;
                    }
                }
            }
        }
    }
    sampled[(size_t)bq * DD + h * HDIM + lane] = f2b(acc);
}

// -------------------------------------------------------------------------
extern "C" void kernel_launch(void* const* d_in, const int* in_sizes, int n_in,
                              void* d_out, int out_size, void* d_ws, size_t ws_size,
                              hipStream_t stream)
{
    (void)in_sizes; (void)n_in; (void)out_size; (void)ws_size;

    const float* query = (const float*)d_in[0];
    const float* refp  = (const float*)d_in[1];
    const float* value = (const float*)d_in[2];
    const float* Wv    = (const float*)d_in[3];
    const float* bv    = (const float*)d_in[4];
    const float* Woff  = (const float*)d_in[5];
    const float* boff  = (const float*)d_in[6];
    const float* Wattn = (const float*)d_in[7];
    const float* battn = (const float*)d_in[8];
    const float* Wout  = (const float*)d_in[9];
    const float* bout  = (const float*)d_in[10];
    // d_in[11] spatial_shapes is compile-time constant here

    char* ws = (char*)d_ws;
    size_t o = 0;
    unsigned short* WvT    = (unsigned short*)(ws + o); o += (size_t)256 * 256 * 2;
    unsigned short* WoffT  = (unsigned short*)(ws + o); o += (size_t)256 * 256 * 2;
    unsigned short* WattnT = (unsigned short*)(ws + o); o += (size_t)128 * 256 * 2;
    unsigned short* WoutT  = (unsigned short*)(ws + o); o += (size_t)256 * 256 * 2;
    unsigned short* vbf    = (unsigned short*)(ws + o); o += (size_t)BB * NHEAD * SS * HDIM * 2;
    float*          off_f  = (float*)(ws + o);          o += (size_t)BB * QQ * 256 * 4;
    float*          logit  = (float*)(ws + o);          o += (size_t)BB * QQ * 128 * 4;
    unsigned short* smp    = (unsigned short*)(ws + o); o += (size_t)BB * QQ * 256 * 2;

    // 1) weights -> transposed bf16
    wcvt_t<<<(256 * 256 + 255) / 256, 256, 0, stream>>>(Wv,    WvT,    256, 256);
    wcvt_t<<<(256 * 256 + 255) / 256, 256, 0, stream>>>(Woff,  WoffT,  256, 256);
    wcvt_t<<<(256 * 128 + 255) / 256, 256, 0, stream>>>(Wattn, WattnT, 256, 128);
    wcvt_t<<<(256 * 256 + 255) / 256, 256, 0, stream>>>(Wout,  WoutT,  256, 256);

    // 2) v = value @ Wv + bv -> bf16 [B,NH,S,HD]
    gemm_bf16<0, false><<<dim3((BB * SS) / 128, 256 / 64), 256, 0, stream>>>(
        value, WvT, bv, vbf, BB * SS, 256, 256, 0);

    // 3) off = query @ Woff + boff ; attn logits = query @ Wattn + battn
    gemm_bf16<1, false><<<dim3((BB * QQ) / 128, 256 / 64), 256, 0, stream>>>(
        query, WoffT, boff, off_f, BB * QQ, 256, 256, 256);
    gemm_bf16<1, false><<<dim3((BB * QQ) / 128, 128 / 64), 256, 0, stream>>>(
        query, WattnT, battn, logit, BB * QQ, 128, 256, 128);

    // 4) fused softmax + deformable bilinear sampling -> bf16 [B*Q, 256]
    msda_sample<<<(BB * QQ * NHEAD) / 8, 256, 0, stream>>>(refp, off_f, logit, vbf, smp);

    // 5) out = sampled @ Wout + bout -> f32 [B*Q, 256]
    gemm_bf16<1, true><<<dim3((BB * QQ) / 128, 256 / 64), 256, 0, stream>>>(
        smp, WoutT, bout, d_out, BB * QQ, 256, 256, 256);
}
